// Interaction_GraphConvolution_51316269252694
// MI455X (gfx1250) — matrix operand order repaired and checked by reading, not compiled, not run on hardware
//
#include <hip/hip_runtime.h>

// ---------------------------------------------------------------------------
// out[a,d] = (1/nc[a]^2) * sum_{i,j} adj[a,i]*mh[i,j]*mf[a,j]*wf[i,d]*wf[j,d]
// wf = node_features @ weight   (N=1024, D_IN=256, D_OUT=128)
//
// ~2.75e11 FLOPs vs ~12MB data -> pure compute-bound; v_wmma_f32_16x16x32_f16
// with f32 accumulation. Each A-tile load feeds TWO wmmas (2 d-tiles/wave) to
// halve A-operand cache traffic per wmma (~250B/cycle/WGP, within WGP$ reach).
// ---------------------------------------------------------------------------

typedef __attribute__((ext_vector_type(16))) _Float16 v16h;
typedef __attribute__((ext_vector_type(8)))  _Float16 v8h;
typedef __attribute__((ext_vector_type(8)))  float    v8f;

#define GN 1024   // N
#define GD 128    // D_OUT
#define KIN 256   // D_IN

// ---------------------------------------------------------------------------
// Kernel 1: mask_hadamard f32 -> f16 (once; main kernel re-reads it 1024x from
// L2, so halving the footprint and removing per-use cvt pays off).
// ---------------------------------------------------------------------------
__global__ void cvt_h_kernel(const float* __restrict__ mh,
                             _Float16* __restrict__ h16) {
    int base = (blockIdx.x * 256 + threadIdx.x) * 4;
#pragma unroll
    for (int e = 0; e < 4; ++e) h16[base + e] = (_Float16)mh[base + e];
}

// ---------------------------------------------------------------------------
// Kernel 2: wf = X @ W via WMMA, stored TRANSPOSED as f16: wfT[d][i].
// Grid: 512 blocks x 1 wave; block = (i-tile, d-tile); K = 256 = 8 chunks.
// (67 MFLOP total -- negligible; strided B loads are fine here.)
// ---------------------------------------------------------------------------
__global__ void wf_gemm_kernel(const float* __restrict__ X,
                               const float* __restrict__ W,
                               _Float16* __restrict__ wfT) {
    const int it    = blockIdx.x >> 3;
    const int dt    = blockIdx.x & 7;
    const int ibase = it * 16;
    const int dbase = dt * 16;
    const int lane  = threadIdx.x & 31;
    const int ln    = lane & 15;
    const int half  = lane >> 4;

    v8f acc = {0.f, 0.f, 0.f, 0.f, 0.f, 0.f, 0.f, 0.f};
#pragma unroll
    for (int kc = 0; kc < 8; ++kc) {
        const int kb = kc * 32;
        // A-operand: X row (ibase+ln); lanes<16 take K {0..7,16..23}, lanes>=16
        // take K {8..15,24..31} (ISA 16-bit A 16x32 layout).
        const float* xr = X + (size_t)(ibase + ln) * KIN + kb + (half ? 8 : 0);
        v16h amat;
#pragma unroll
        for (int e = 0; e < 8; ++e) {
            amat[e]     = (_Float16)xr[e];
            amat[8 + e] = (_Float16)xr[16 + e];
        }
        // B-operand: column d = dbase+ln, same K split as A (strided in W).
        const float* wp = W + (size_t)(kb + (half ? 8 : 0)) * GD + dbase + ln;
        v16h bmat;
#pragma unroll
        for (int e = 0; e < 16; ++e) bmat[e] = (_Float16)wp[(size_t)(e < 8 ? e : e + 8) * GD];

        acc = __builtin_amdgcn_wmma_f32_16x16x32_f16(
            false, amat, false, bmat, (short)0, acc, false, false);
    }
    // C/D layout: VGPR r -> M = r + 8*half, N = ln; rows consecutive in r, so
    // the transposed store is one contiguous v8h (16B) store per lane.
    v8h o;
#pragma unroll
    for (int r = 0; r < 8; ++r) o[r] = (_Float16)acc[r];
    *(v8h*)(&wfT[(size_t)(dbase + ln) * GN + ibase + (half ? 8 : 0)]) = o;
}

// ---------------------------------------------------------------------------
// Kernel 3: main contraction. One WG per output row a; 8 waves.
// Wave w: i-half = w>>2 (i-tiles [32*ih, 32*ih+32)), d-tile pair = w&3
// (d-tiles at 32*(w&3) and 32*(w&3)+16). Each A-tile load feeds 2 wmmas.
//
//  * B operands (mf .* wf per j-chunk) are i-tile-invariant -> hoist a block
//    of 8 prescaled B operands per d-tile (2x64 VGPRs) out of the i-loop.
//  * the i-fold is linear -> fold PARTIAL C tiles (8 j-chunks), so no
//    long-lived accumulators and any loop order is legal.
//  * the two i-halves are combined through a 1KB LDS buffer at the end.
// ---------------------------------------------------------------------------
__global__ __launch_bounds__(256) void graphconv_main_kernel(
    const float* __restrict__ adj, const float* __restrict__ mf,
    const float* __restrict__ ncount, const _Float16* __restrict__ h16,
    const _Float16* __restrict__ wfT, float* __restrict__ out) {

    __shared__ __attribute__((aligned(32))) _Float16 sM[GN];   // 2KB
    __shared__ float sPart[2][GD];                             // 1KB

    const int a      = blockIdx.x;
    const int tid    = threadIdx.x;
    const int wave   = tid >> 5;
    const int lane   = tid & 31;
    const int ln     = lane & 15;
    const int half   = lane >> 4;
    const int ih     = wave >> 2;          // i-half: 0 or 1
    const int dbase0 = (wave & 3) * 32;    // first of two d-tiles
    const int itlo   = ih * 32;            // this wave's i-tile range

    // mf[a,:] -> f16 in LDS (feeds packed-f16 B-operand scaling)
    for (int j = tid; j < GN; j += 256)
        sM[j] = (_Float16)mf[(size_t)a * GN + j];
    __syncthreads();

    const _Float16* wfrow0 = wfT + (size_t)(dbase0 + ln) * GN;       // d-tile 0
    const _Float16* wfrow1 = wfT + (size_t)(dbase0 + 16 + ln) * GN;  // d-tile 1
    const float*    arow0  = adj + (size_t)a * GN;
    float pacc0 = 0.f, pacc1 = 0.f;

#pragma unroll 1
    for (int jb = 0; jb < 4; ++jb) {           // blocks of 8 j-chunks
        // ---- build 2x8 prescaled B operands: Z[j,d] = mf[a,j]*wf[j,d] ----
        v16h B0[8], B1[8];
#pragma unroll
        for (int q = 0; q < 8; ++q) {
            const int bo = (jb * 8 + q) * 32 + (half ? 16 : 0);
            v16h bm = *(const v16h*)(&sM[bo]);               // mf[a,j]
            B0[q] = *(const v16h*)(wfrow0 + bo) * bm;        // v_pk_mul_f16 x8
            B1[q] = *(const v16h*)(wfrow1 + bo) * bm;
        }

#pragma unroll 1
        for (int t = 0; t < 32; ++t) {         // this wave's i-tiles
            const int it = itlo + t;
            // A-operand base: row (it*16 + ln) of h16; lanes<16 read K
            // {0..15}, lanes>=16 read {16..31} of each j-chunk.
            const _Float16* hp =
                h16 + (size_t)(it * 16 + ln) * GN + (half ? 16 : 0);

            v8f acc0 = {0.f, 0.f, 0.f, 0.f, 0.f, 0.f, 0.f, 0.f};
            v8f acc1 = {0.f, 0.f, 0.f, 0.f, 0.f, 0.f, 0.f, 0.f};
#pragma unroll
            for (int q = 0; q < 8; ++q) {
                const int ao = (jb * 8 + q) * 32;
                v8h a0 = *(const v8h*)(hp + ao);
                v8h a1 = *(const v8h*)(hp + ao + 8);
                v16h amat = __builtin_shufflevector(
                    a0, a1, 0, 1, 2, 3, 4, 5, 6, 7,
                    8, 9, 10, 11, 12, 13, 14, 15);
                // one A load feeds two wmmas (independent accumulators)
                acc0 = __builtin_amdgcn_wmma_f32_16x16x32_f16(
                    false, amat, false, B0[q], (short)0, acc0, false, false);
                acc1 = __builtin_amdgcn_wmma_f32_16x16x32_f16(
                    false, amat, false, B1[q], (short)0, acc1, false, false);
            }

            // ---- fold partial C tiles: C[M=r+8*half, ln]*wf[i,d]*adj[a,i]
            const int ibase = it * 16 + (half ? 8 : 0);
            v8h wfv0 = *(const v8h*)(wfrow0 + ibase);  // wf[i,d0], i contig
            v8h wfv1 = *(const v8h*)(wfrow1 + ibase);  // wf[i,d1]
            const float* arow = arow0 + ibase;         // adj[a,i]
#pragma unroll
            for (int r = 0; r < 8; ++r) {
                const float av = arow[r];
                pacc0 += acc0[r] * (float)wfv0[r] * av;
                pacc1 += acc1[r] * (float)wfv1[r] * av;
            }

            if (t < 31)  // pull next i-tile's A rows toward the WGP
                __builtin_prefetch(hp + 16 * GN, 0, 1);
        }
    }

    // lanes l and l+16 hold the same d (rows 0-7 vs 8-15): combine halves.
    const float tot0 = pacc0 + __shfl_xor(pacc0, 16, 32);
    const float tot1 = pacc1 + __shfl_xor(pacc1, 16, 32);
    if (lane < 16) {
        sPart[ih][dbase0 + lane]      = tot0;
        sPart[ih][dbase0 + 16 + lane] = tot1;
    }
    __syncthreads();

    // combine the two i-halves and write out
    if (tid < GD) {
        const float nv = ncount[a];
        out[(size_t)a * GD + tid] =
            (sPart[0][tid] + sPart[1][tid]) / (nv * nv);
    }
}

// ---------------------------------------------------------------------------
extern "C" void kernel_launch(void* const* d_in, const int* in_sizes, int n_in,
                              void* d_out, int out_size, void* d_ws, size_t ws_size,
                              hipStream_t stream) {
    (void)in_sizes; (void)n_in; (void)out_size; (void)ws_size;
    const float* nf  = (const float*)d_in[0];  // node_features  [1024,256]
    const float* adj = (const float*)d_in[1];  // adjacency      [1024,1024]
    const float* mf  = (const float*)d_in[2];  // mask_father    [1024,1,1024]
    const float* nc  = (const float*)d_in[3];  // neighbor_count [1024,1]
    const float* mh  = (const float*)d_in[4];  // mask_hadamard  [1024,1,1024]
    const float* wt  = (const float*)d_in[5];  // weight         [256,128]
    float* out = (float*)d_out;                // [1024,128]

    _Float16* wfT = (_Float16*)d_ws;                        // 256 KB
    _Float16* h16 = (_Float16*)((char*)d_ws + 262144);      // 2 MB

    cvt_h_kernel<<<1024, 256, 0, stream>>>(mh, h16);
    wf_gemm_kernel<<<512, 32, 0, stream>>>(nf, wt, wfT);
    graphconv_main_kernel<<<1024, 256, 0, stream>>>(adj, mf, nc, h16, wfT, out);
}
